// DecoderRNN_83562883711169
// MI455X (gfx1250) — matrix operand order repaired and checked by reading, not compiled, not run on hardware
//
#include <hip/hip_runtime.h>
#include <hip/hip_bf16.h>

typedef __attribute__((ext_vector_type(16))) _Float16 v16h;
typedef __attribute__((ext_vector_type(8)))  float    v8f;

#define B_   128
#define T_   33
#define L_   32
#define D_   512
#define V_   10000
#define G4_  2048   // 4*D

// ---------------------------------------------------------------------------
// Fragment loaders per CDNA5 ISA 7.12.2 wave32 layouts (f16, 16x16x32).
// ---------------------------------------------------------------------------
union Frag { v16h v; unsigned u[8]; };

__device__ __forceinline__ Frag load_a(const _Float16* __restrict__ arow, int k0, int hi)
{
    Frag a;
    #pragma unroll
    for (int j = 0; j < 4; ++j) {
        a.u[j]     = *(const unsigned*)(arow + k0 + 2*j + hi*8);
        a.u[j + 4] = *(const unsigned*)(arow + k0 + 16 + 2*j + hi*8);
    }
    return a;
}

__device__ __forceinline__ Frag load_b(const _Float16* __restrict__ brow, int k0, int hi)
{
    Frag b;
    #pragma unroll
    for (int j = 0; j < 4; ++j) {
        b.u[j]     = *(const unsigned*)(brow + k0 + 2*j + hi*8);
        b.u[j + 4] = *(const unsigned*)(brow + k0 + 16 + 2*j + hi*8);
    }
    return b;
}

// ---------------------------------------------------------------------------
// Single-tile GEMM (used for the latency-bound recurrent h @ Whh.T):
// C[m,n] = sum_k A[m,k]*Bt[n,k] (+ Cinit[m,n])
// ---------------------------------------------------------------------------
__global__ __launch_bounds__(256) void gemm_tn(
        const _Float16* __restrict__ A, const _Float16* __restrict__ Bt,
        const float* __restrict__ Cinit, float* __restrict__ C,
        int N, int K, int tiles_n, int total_tiles)
{
    const int wid  = threadIdx.x >> 5;
    const int lane = threadIdx.x & 31;
    const int gw   = blockIdx.x * 8 + wid;
    if (gw >= total_tiles) return;              // wave-uniform: EXEC stays all-1
    const int m0 = (gw / tiles_n) * 16;
    const int n0 = (gw % tiles_n) * 16;
    const int hl = lane & 15, hi = lane >> 4;

    union { v8f v; float f[8]; } acc;
    if (Cinit) {
        #pragma unroll
        for (int r = 0; r < 8; ++r)
            acc.f[r] = Cinit[(size_t)(m0 + r + hi*8) * N + n0 + hl];
    } else {
        #pragma unroll
        for (int r = 0; r < 8; ++r) acc.f[r] = 0.0f;
    }

    const _Float16* arow = A  + (size_t)(m0 + hl) * K;
    const _Float16* brow = Bt + (size_t)(n0 + hl) * K;
    for (int k0 = 0; k0 < K; k0 += 32) {
        Frag a = load_a(arow, k0, hi);
        Frag b = load_b(brow, k0, hi);
        acc.v = __builtin_amdgcn_wmma_f32_16x16x32_f16(false, a.v, false, b.v,
                                                       (short)0, acc.v, false, false);
    }

    #pragma unroll
    for (int r = 0; r < 8; ++r)
        C[(size_t)(m0 + r + hi*8) * N + n0 + hl] = acc.f[r];
}

// ---------------------------------------------------------------------------
// 4x M-blocked GEMM: each wave owns 4 M-tiles x 1 N-tile, reusing the B
// (weight) fragment across 4 WMMAs per k-step (10 dwords loaded per WMMA).
// MODE 0: C[m*N+n] = acc + bias0[n] + bias1[n]        (Xih precompute)
// MODE 1: out[remap(m)*V + n] = acc + bias0[n]        (FC with (t,b)->(b,l) remap)
// ---------------------------------------------------------------------------
template<int MODE>
__global__ __launch_bounds__(256) void gemm_m4(
        const _Float16* __restrict__ A, const _Float16* __restrict__ Bt,
        const float* __restrict__ bias0, const float* __restrict__ bias1,
        float* __restrict__ C, int N, int K, int tiles_n, int total_groups)
{
    const int wid  = threadIdx.x >> 5;
    const int lane = threadIdx.x & 31;
    const int gw   = blockIdx.x * 8 + wid;
    if (gw >= total_groups) return;             // wave-uniform
    const int m0 = (gw / tiles_n) * 64;         // 4 M-tiles
    const int n0 = (gw % tiles_n) * 16;
    const int hl = lane & 15, hi = lane >> 4;

    union { v8f v; float f[8]; } acc[4];
    #pragma unroll
    for (int i = 0; i < 4; ++i)
        #pragma unroll
        for (int r = 0; r < 8; ++r) acc[i].f[r] = 0.0f;

    const _Float16* brow  = Bt + (size_t)(n0 + hl) * K;
    const _Float16* arow0 = A  + (size_t)(m0 + hl) * K;
    const size_t astride = (size_t)16 * K;

    for (int k0 = 0; k0 < K; k0 += 32) {
        Frag b = load_b(brow, k0, hi);
        #pragma unroll
        for (int i = 0; i < 4; ++i) {
            Frag a = load_a(arow0 + (size_t)i * astride, k0, hi);
            acc[i].v = __builtin_amdgcn_wmma_f32_16x16x32_f16(false, a.v, false, b.v,
                                                              (short)0, acc[i].v, false, false);
        }
    }

    if (MODE == 0) {
        float badd = bias0[n0 + hl] + bias1[n0 + hl];
        #pragma unroll
        for (int i = 0; i < 4; ++i)
            #pragma unroll
            for (int r = 0; r < 8; ++r)
                C[(size_t)(m0 + i*16 + r + hi*8) * N + n0 + hl] = acc[i].f[r] + badd;
    } else {
        const float bv = bias0[n0 + hl];
        #pragma unroll
        for (int i = 0; i < 4; ++i)
            #pragma unroll
            for (int r = 0; r < 8; ++r) {
                const int m    = m0 + i*16 + r + hi*8;      // m = l*128 + b
                const int orow = (m & 127) * L_ + (m >> 7); // -> b*32 + l
                C[(size_t)orow * V_ + n0 + hl] = acc[i].f[r] + bv;
            }
    }
}

// ---------------------------------------------------------------------------
// Build model input [T=33][B=128][D=512] in f16: t=0 -> features, t>0 -> embedding
// ---------------------------------------------------------------------------
__global__ void prep_inp(const float* __restrict__ feat, const int* __restrict__ cap,
                         const float* __restrict__ embW, _Float16* __restrict__ inpH)
{
    const int idx = blockIdx.x * blockDim.x + threadIdx.x;
    if (idx >= T_ * B_ * D_) return;
    const int k = idx & (D_ - 1);
    const int m = idx / D_;
    const int t = m / B_, b = m % B_;
    float v;
    if (t == 0) v = feat[(size_t)b * D_ + k];
    else        v = embW[(size_t)cap[b * L_ + (t - 1)] * D_ + k];
    inpH[idx] = (_Float16)v;
}

__global__ void cvt_f2h(const float* __restrict__ in, _Float16* __restrict__ out, int n)
{
    const int i = blockIdx.x * blockDim.x + threadIdx.x;
    if (i < n) out[i] = (_Float16)in[i];
}

// ---------------------------------------------------------------------------
// LSTM cell elementwise: gates G[128][2048] (i|f|g|o), updates c, emits h (f32+f16)
// ---------------------------------------------------------------------------
__global__ void lstm_cell(const float* __restrict__ G, float* __restrict__ c,
                          float* __restrict__ hF, _Float16* __restrict__ hH)
{
    const int idx = blockIdx.x * blockDim.x + threadIdx.x;
    if (idx >= B_ * D_) return;
    const int b = idx / D_, j = idx & (D_ - 1);
    const float* g = G + (size_t)b * G4_;
    const float gi = g[j], gf = g[D_ + j], gg = g[2 * D_ + j], go = g[3 * D_ + j];
    const float si = 1.0f / (1.0f + __expf(-gi));
    const float sf = 1.0f / (1.0f + __expf(-gf));
    const float so = 1.0f / (1.0f + __expf(-go));
    const float cn = sf * c[idx] + si * tanhf(gg);
    const float hn = so * tanhf(cn);
    c[idx]  = cn;
    hF[idx] = hn;
    hH[idx] = (_Float16)hn;
}

// ---------------------------------------------------------------------------
// LayerNorm over D=512 per row; writes f16 h into Hseq[t]
// ---------------------------------------------------------------------------
__global__ __launch_bounds__(256) void layernorm(const float* __restrict__ h,
                                                 const float* __restrict__ g,
                                                 const float* __restrict__ bb,
                                                 _Float16* __restrict__ outH)
{
    __shared__ float sdata[256];
    const int row = blockIdx.x, tid = threadIdx.x;
    const float x0 = h[row * D_ + tid];
    const float x1 = h[row * D_ + tid + 256];
    sdata[tid] = x0 + x1;
    __syncthreads();
    for (int s = 128; s > 0; s >>= 1) {
        if (tid < s) sdata[tid] += sdata[tid + s];
        __syncthreads();
    }
    const float mu = sdata[0] * (1.0f / D_);
    __syncthreads();
    const float d0 = x0 - mu, d1 = x1 - mu;
    sdata[tid] = d0 * d0 + d1 * d1;
    __syncthreads();
    for (int s = 128; s > 0; s >>= 1) {
        if (tid < s) sdata[tid] += sdata[tid + s];
        __syncthreads();
    }
    const float rstd = rsqrtf(sdata[0] * (1.0f / D_) + 1e-5f);
    outH[row * D_ + tid]       = (_Float16)(d0 * rstd * g[tid]       + bb[tid]);
    outH[row * D_ + tid + 256] = (_Float16)(d1 * rstd * g[tid + 256] + bb[tid + 256]);
}

// ---------------------------------------------------------------------------
// In-place log-softmax over V=10000 per row of d_out
// ---------------------------------------------------------------------------
__global__ __launch_bounds__(256) void log_softmax(float* __restrict__ out)
{
    __shared__ float sdata[256];
    float* row = out + (size_t)blockIdx.x * V_;
    const int tid = threadIdx.x;
    float m = -3.4e38f;
    for (int i = tid; i < V_; i += 256) m = fmaxf(m, row[i]);
    sdata[tid] = m;
    __syncthreads();
    for (int s = 128; s > 0; s >>= 1) {
        if (tid < s) sdata[tid] = fmaxf(sdata[tid], sdata[tid + s]);
        __syncthreads();
    }
    m = sdata[0];
    __syncthreads();
    float acc = 0.0f;
    for (int i = tid; i < V_; i += 256) acc += __expf(row[i] - m);
    sdata[tid] = acc;
    __syncthreads();
    for (int s = 128; s > 0; s >>= 1) {
        if (tid < s) sdata[tid] += sdata[tid + s];
        __syncthreads();
    }
    const float lse = m + __logf(sdata[0]);
    for (int i = tid; i < V_; i += 256) row[i] = row[i] - lse;
}

// ---------------------------------------------------------------------------
extern "C" void kernel_launch(void* const* d_in, const int* in_sizes, int n_in,
                              void* d_out, int out_size, void* d_ws, size_t ws_size,
                              hipStream_t stream)
{
    const float* features = (const float*)d_in[0];
    const int*   caption  = (const int*)  d_in[1];
    const float* emb_W    = (const float*)d_in[2];
    const float* W_ih     = (const float*)d_in[3];
    const float* W_hh     = (const float*)d_in[4];
    const float* b_ih     = (const float*)d_in[5];
    const float* b_hh     = (const float*)d_in[6];
    const float* ln_g     = (const float*)d_in[7];
    const float* ln_b     = (const float*)d_in[8];
    const float* fc_W     = (const float*)d_in[9];
    const float* fc_b     = (const float*)d_in[10];
    float* out = (float*)d_out;

    // -------- workspace carve-up (256B aligned) --------
    char* ws = (char*)d_ws;
    size_t off = 0;
    auto take = [&](size_t bytes) { char* p = ws + off; off = (off + bytes + 255) & ~(size_t)255; return p; };
    _Float16* inpH  = (_Float16*)take((size_t)T_ * B_ * D_ * 2);      // [33][128][512]
    _Float16* WihH  = (_Float16*)take((size_t)2 * G4_ * D_ * 2);
    _Float16* WhhH  = (_Float16*)take((size_t)2 * G4_ * D_ * 2);
    _Float16* fcWH  = (_Float16*)take((size_t)V_ * D_ * 2);
    float*    Xih   = (float*)   take((size_t)2 * T_ * B_ * G4_ * 4); // precomputed x@Wih.T+b
    float*    G     = (float*)   take((size_t)B_ * G4_ * 4);
    float*    cbuf  = (float*)   take((size_t)B_ * D_ * 4);
    float*    hF    = (float*)   take((size_t)B_ * D_ * 4);
    _Float16* hHtmp = (_Float16*)take((size_t)B_ * D_ * 2);
    _Float16* zeroH = (_Float16*)take((size_t)B_ * D_ * 2);
    _Float16* HseqH = (_Float16*)take((size_t)T_ * B_ * D_ * 2);      // LN'd h per step (f16)

    // -------- per-call state init (graph-capture safe) --------
    (void)hipMemsetAsync(cbuf,  0, (size_t)B_ * D_ * 4, stream);
    (void)hipMemsetAsync(zeroH, 0, (size_t)B_ * D_ * 2, stream);

    // -------- input assembly + f16 weight conversion --------
    prep_inp<<<(T_ * B_ * D_ + 255) / 256, 256, 0, stream>>>(features, caption, emb_W, inpH);
    cvt_f2h<<<(2 * G4_ * D_ + 255) / 256, 256, 0, stream>>>(W_ih, WihH, 2 * G4_ * D_);
    cvt_f2h<<<(2 * G4_ * D_ + 255) / 256, 256, 0, stream>>>(W_hh, WhhH, 2 * G4_ * D_);
    cvt_f2h<<<(V_ * D_ + 255) / 256, 256, 0, stream>>>(fc_W, fcWH, V_ * D_);

    // -------- batched precompute: Xih[l][t][b][n] = x @ Wih[l].T + b_ih + b_hh --------
    {
        const int tiles_n = G4_ / 16;                       // 128
        const int groups  = (T_ * B_ / 64) * tiles_n;       // 66*128 = 8448
        const int blocks  = (groups + 7) / 8;
        for (int l = 0; l < 2; ++l)
            gemm_m4<0><<<blocks, 256, 0, stream>>>(
                inpH, WihH + (size_t)l * G4_ * D_,
                b_ih + l * G4_, b_hh + l * G4_,
                Xih + (size_t)l * T_ * B_ * G4_,
                G4_, D_, tiles_n, groups);
    }

    // -------- sequential recurrence --------
    {
        const int tiles_n = G4_ / 16;                  // 128
        const int total   = (B_ / 16) * tiles_n;       // 8*128 = 1024
        const int blocks  = (total + 7) / 8;           // 128
        for (int t = 0; t < T_; ++t) {
            const _Float16* hprev = (t == 0) ? zeroH : (HseqH + (size_t)(t - 1) * B_ * D_);
            // layer 0: G = Xih0[t] + hprev @ Whh0.T ; cell
            gemm_tn<<<blocks, 256, 0, stream>>>(
                hprev, WhhH, Xih + (size_t)t * B_ * G4_,
                G, G4_, D_, tiles_n, total);
            lstm_cell<<<(B_ * D_ + 255) / 256, 256, 0, stream>>>(G, cbuf, hF, hHtmp);
            // layer 1: G = Xih1[t] + h0 @ Whh1.T ; cell
            gemm_tn<<<blocks, 256, 0, stream>>>(
                hHtmp, WhhH + (size_t)G4_ * D_,
                Xih + ((size_t)T_ + t) * B_ * G4_,
                G, G4_, D_, tiles_n, total);
            lstm_cell<<<(B_ * D_ + 255) / 256, 256, 0, stream>>>(G, cbuf, hF, hHtmp);
            // layernorm -> f16 carry / FC input
            layernorm<<<B_, 256, 0, stream>>>(hF, ln_g, ln_b, HseqH + (size_t)t * B_ * D_);
        }
    }

    // -------- batched vocab projection (rows t=1..32), remapped into d_out --------
    {
        const int tiles_n = V_ / 16;                       // 625
        const int groups  = (B_ * L_ / 64) * tiles_n;      // 64*625 = 40000
        const int blocks  = (groups + 7) / 8;
        gemm_m4<1><<<blocks, 256, 0, stream>>>(
            HseqH + (size_t)B_ * D_, fcWH, fc_b, nullptr, out, V_, D_, tiles_n, groups);
    }

    // -------- in-place log-softmax over V per output row --------
    log_softmax<<<B_ * L_, 256, 0, stream>>>(out);
}
